// MoEMLP_42348377538843
// MI455X (gfx1250) — matrix operand order repaired and checked by reading, not compiled, not run on hardware
//
#include <hip/hip_runtime.h>
#include <hip/hip_bf16.h>
#include <math.h>

// ---------------------------------------------------------------------------
// MoE MLP for MI455X (gfx1250): bf16 WMMA, LDS-staged weights, fused experts.
// ---------------------------------------------------------------------------

typedef __attribute__((ext_vector_type(16))) __bf16 v16bf;
typedef __attribute__((ext_vector_type(8)))  __bf16 v8bf;
typedef __attribute__((ext_vector_type(8)))  float  v8f;
typedef __attribute__((ext_vector_type(4)))  float  v4f;

#define D_DIM   1024
#define F_DIM   4096
#define E_NUM   8
#define TILE_T  32     // tokens per workgroup
#define FCHUNK  32     // F columns processed per iteration (= WMMA K)

// LDS strides (in bf16 halves); all chosen so fragment loads are 16B aligned.
#define XS   1032      // xbuf  [32 tokens][1024 d]   row stride
#define WIS  1032      // winT  [32 f][1024 d]        row stride (transposed W_in chunk)
#define WOS  40        // woutT [1024 d][32 f]        row stride (transposed W_out chunk)
#define HS   40        // hbuf  [32 tokens][32 f]     row stride

#define XBUF_OFF   0
#define UBUF_OFF   (TILE_T * XS)                       // 33024 halves
#define HBUF_OFF   (UBUF_OFF + D_DIM * WOS)            // 33024 + 40960 = 73984 halves
#define LDS_HALVES (HBUF_OFF + TILE_T * HS)            // 75264 halves = 150528 bytes

// Fragment loaders per ISA 7.12.2 layouts (wave32):
//  A (16x32 bf16): halves 0-7 -> K = base+8*hi .. +7 ; halves 8-15 -> K = base+16+8*hi .. +7
//  B (32x16 bf16): halves 0-7 -> K = base+8*hi .. +7 ; halves 8-15 -> K = base+16+8*hi .. +7 ; N = lane&15 (row of transposed buf)
static __device__ __forceinline__ v16bf load_afrag(const __bf16* row, int kk, int hi) {
  v16bf a;
  ((v8bf*)&a)[0] = *(const v8bf*)(row + kk + 8 * hi);
  ((v8bf*)&a)[1] = *(const v8bf*)(row + kk + 16 + 8 * hi);
  return a;
}
static __device__ __forceinline__ v16bf load_bfrag(const __bf16* row, int kk, int hi) {
  v16bf b;
  ((v8bf*)&b)[0] = *(const v8bf*)(row + kk + 8 * hi);
  ((v8bf*)&b)[1] = *(const v8bf*)(row + kk + 16 + 8 * hi);
  return b;
}

// ---------------------------------------------------------------------------
// Router: one wave32 per token. logits = x . W_router, softmax, top-2,
// renormalize -> dense combine[T, E] (zeros for unselected experts).
// ---------------------------------------------------------------------------
__global__ __launch_bounds__(256) void moe_router(const float* __restrict__ x,
                                                  const float* __restrict__ wr,
                                                  float* __restrict__ comb, int T) {
  const int wave = threadIdx.x >> 5;
  const int lane = threadIdx.x & 31;
  const int t = blockIdx.x * 8 + wave;
  if (t >= T) return;

  float acc[E_NUM];
#pragma unroll
  for (int e = 0; e < E_NUM; ++e) acc[e] = 0.f;

  const float* xr = x + (size_t)t * D_DIM;
  for (int d = lane * 4; d < D_DIM; d += 128) {
    v4f xv = *(const v4f*)(xr + d);
#pragma unroll
    for (int e = 0; e < E_NUM; ++e) {
      v4f wv = *(const v4f*)(wr + e * D_DIM + d);
      acc[e] += xv.x * wv.x + xv.y * wv.y + xv.z * wv.z + xv.w * wv.w;
    }
  }
#pragma unroll
  for (int e = 0; e < E_NUM; ++e)
#pragma unroll
    for (int off = 16; off > 0; off >>= 1)
      acc[e] += __shfl_down(acc[e], off, 32);

  if (lane == 0) {
    float m = acc[0];
#pragma unroll
    for (int e = 1; e < E_NUM; ++e) m = fmaxf(m, acc[e]);
    float p[E_NUM];
#pragma unroll
    for (int e = 0; e < E_NUM; ++e) p[e] = __expf(acc[e] - m);
    // top-2 (softmax denominator cancels in the renormalization)
    int i1 = 0; float p1 = p[0];
#pragma unroll
    for (int e = 1; e < E_NUM; ++e) if (p[e] > p1) { p1 = p[e]; i1 = e; }
    int i2 = -1; float p2 = -1.f;
#pragma unroll
    for (int e = 0; e < E_NUM; ++e) if (e != i1 && p[e] > p2) { p2 = p[e]; i2 = e; }
    float inv = 1.f / (p1 + p2);
#pragma unroll
    for (int e = 0; e < E_NUM; ++e) {
      float c = (e == i1) ? p1 * inv : ((e == i2) ? p2 * inv : 0.f);
      comb[(size_t)t * E_NUM + e] = c;
    }
  }
}

// ---------------------------------------------------------------------------
// Expert kernel: grid (T/32, E). 256 threads = 8 wave32.
// y[tile] += combine[t,e] * (GELU(x W_in[e] + b_in) W_out[e] + b_out)
// ---------------------------------------------------------------------------
__global__ __launch_bounds__(256) void moe_expert(const float* __restrict__ x,
                                                  const float* __restrict__ Win,
                                                  const float* __restrict__ bin,
                                                  const float* __restrict__ Wout,
                                                  const float* __restrict__ bout,
                                                  const float* __restrict__ comb,
                                                  float* __restrict__ out) {
  extern __shared__ __align__(16) __bf16 lds[];
  __bf16* xbuf = lds + XBUF_OFF;   // [32][XS]  x tile, bf16
  __bf16* ubuf = lds + UBUF_OFF;   // union: winT [32][WIS]  /  woutT [1024][WOS]
  __bf16* hbuf = lds + HBUF_OFF;   // [32][HS]  GELU(h) tile, bf16

  const int e     = blockIdx.y;
  const int tbase = blockIdx.x * TILE_T;
  const int tid   = threadIdx.x;
  const int lane  = tid & 31;
  const int wave  = tid >> 5;
  const int hi    = (lane >> 4) & 1;   // lane >= 16
  const int l15   = lane & 15;

  // Early out: skip tile entirely if no token here routed to this expert.
  float anyc = 0.f;
  for (int i = 0; i < TILE_T; ++i) anyc += comb[(size_t)(tbase + i) * E_NUM + e];
  if (anyc == 0.f) return;

  const float* WinE  = Win  + (size_t)e * D_DIM * F_DIM;
  const float* WoutE = Wout + (size_t)e * F_DIM * D_DIM;

  // ---- Stage X tile -> bf16 LDS (row-major [token][d]) ----
  {
    const int row = tid >> 3;
    const int dg  = (tid & 7) * 4;
    const float* xr = x + (size_t)(tbase + row) * D_DIM;
#pragma unroll 4
    for (int it = 0; it < 32; ++it) {
      int d = dg + it * 32;
      v4f v = *(const v4f*)(xr + d);
      __bf16* p = &xbuf[row * XS + d];
      p[0] = (__bf16)v.x; p[1] = (__bf16)v.y; p[2] = (__bf16)v.z; p[3] = (__bf16)v.w;
    }
  }
  __syncthreads();

  // GEMM1 tile assignment (waves 0-3): h tile (mg1*16 rows, fh1*16 f-cols)
  const int mg1 = wave >> 1;
  const int fh1 = wave & 1;
  // GEMM2 tile assignment (all 8 waves): rows mg2*16, d-cols nq*256..+255
  const int mg2 = wave & 1;
  const int nq  = wave >> 1;

  v8f acc[16];                      // y accumulators: 16 n-tiles x 8 f32
#pragma unroll
  for (int i = 0; i < 16; ++i)
#pragma unroll
    for (int j = 0; j < 8; ++j) acc[i][j] = 0.f;

  for (int f0 = 0; f0 < F_DIM; f0 += FCHUNK) {
    // Prefetch next iteration's weight chunks into cache (global_prefetch_b8).
    if (f0 + FCHUNK < F_DIM) {
#pragma unroll
      for (int r = 0; r < 4; ++r) {
        int q = tid * 4 + r;                               // 0..1023
        __builtin_prefetch(WinE  + (size_t)q * F_DIM + f0 + FCHUNK, 0, 1);
        __builtin_prefetch(WoutE + (size_t)(f0 + FCHUNK + (q >> 5)) * D_DIM
                                 + (q & 31) * 32, 0, 1);
      }
    }

    // ---- Stage W_in[:, f0:f0+32] transposed -> winT [f'][d] (bf16) ----
    {
      const int jg = (tid & 7) * 4;
      const int d0 = tid >> 3;
#pragma unroll 4
      for (int it = 0; it < 32; ++it) {
        int d = d0 + it * 32;
        v4f v = *(const v4f*)(WinE + (size_t)d * F_DIM + f0 + jg);
        ubuf[(jg + 0) * WIS + d] = (__bf16)v.x;
        ubuf[(jg + 1) * WIS + d] = (__bf16)v.y;
        ubuf[(jg + 2) * WIS + d] = (__bf16)v.z;
        ubuf[(jg + 3) * WIS + d] = (__bf16)v.w;
      }
    }
    __syncthreads();

    // ---- GEMM1 (waves 0-3): h = X @ W_in chunk, K = 1024, pipelined ----
    v8f hacc;
#pragma unroll
    for (int j = 0; j < 8; ++j) hacc[j] = 0.f;
    if (wave < 4) {
      const __bf16* arow = &xbuf[(mg1 * 16 + l15) * XS];
      const __bf16* brow = &ubuf[(fh1 * 16 + l15) * WIS];
      // software pipeline: load fragments for kk+32 while WMMA for kk runs
      v16bf a0 = load_afrag(arow, 0, hi);
      v16bf b0 = load_bfrag(brow, 0, hi);
#pragma unroll 4
      for (int kk = 32; kk < D_DIM; kk += 32) {
        v16bf a1 = load_afrag(arow, kk, hi);
        v16bf b1 = load_bfrag(brow, kk, hi);
        hacc = __builtin_amdgcn_wmma_f32_16x16x32_bf16(false, a0, false, b0,
                                                       (short)0, hacc, false, false);
        a0 = a1; b0 = b1;
      }
      hacc = __builtin_amdgcn_wmma_f32_16x16x32_bf16(false, a0, false, b0,
                                                     (short)0, hacc, false, false);
    }
    __syncthreads();

    // ---- bias + exact GELU -> hbuf (waves 0-3) ----
    if (wave < 4) {
      float bia = bin[(size_t)e * F_DIM + f0 + fh1 * 16 + l15];
#pragma unroll
      for (int v = 0; v < 8; ++v) {
        float hv = hacc[v] + bia;
        hv = 0.5f * hv * (1.0f + erff(hv * 0.70710678118654752440f));
        hbuf[(mg1 * 16 + v + 8 * hi) * HS + fh1 * 16 + l15] = (__bf16)hv;
      }
    }

    // ---- Stage W_out[f0:f0+32, :] transposed -> woutT [d][f'] (all threads) ----
    {
      const int j0 = wave * 4;
#pragma unroll
      for (int r = 0; r < 4; ++r) {
        const float* wrow = WoutE + (size_t)(f0 + j0 + r) * D_DIM;
#pragma unroll
        for (int s = 0; s < 8; ++s) {
          int d = lane * 4 + s * 128;
          v4f v = *(const v4f*)(wrow + d);
          ubuf[(size_t)(d + 0) * WOS + j0 + r] = (__bf16)v.x;
          ubuf[(size_t)(d + 1) * WOS + j0 + r] = (__bf16)v.y;
          ubuf[(size_t)(d + 2) * WOS + j0 + r] = (__bf16)v.z;
          ubuf[(size_t)(d + 3) * WOS + j0 + r] = (__bf16)v.w;
        }
      }
    }
    __syncthreads();

    // ---- GEMM2 (all 8 waves): acc += h @ W_out chunk, K = 32 ----
    // B fragments loaded in groups of 4 so ds_load latency overlaps WMMA issue.
    {
      const __bf16* arow = &hbuf[(mg2 * 16 + l15) * HS];
      v16bf a = load_afrag(arow, 0, hi);
#pragma unroll
      for (int g = 0; g < 4; ++g) {
        v16bf b0 = load_bfrag(&ubuf[(size_t)(nq * 256 + (g * 4 + 0) * 16 + l15) * WOS], 0, hi);
        v16bf b1 = load_bfrag(&ubuf[(size_t)(nq * 256 + (g * 4 + 1) * 16 + l15) * WOS], 0, hi);
        v16bf b2 = load_bfrag(&ubuf[(size_t)(nq * 256 + (g * 4 + 2) * 16 + l15) * WOS], 0, hi);
        v16bf b3 = load_bfrag(&ubuf[(size_t)(nq * 256 + (g * 4 + 3) * 16 + l15) * WOS], 0, hi);
        acc[g * 4 + 0] = __builtin_amdgcn_wmma_f32_16x16x32_bf16(false, a, false, b0,
                                                 (short)0, acc[g * 4 + 0], false, false);
        acc[g * 4 + 1] = __builtin_amdgcn_wmma_f32_16x16x32_bf16(false, a, false, b1,
                                                 (short)0, acc[g * 4 + 1], false, false);
        acc[g * 4 + 2] = __builtin_amdgcn_wmma_f32_16x16x32_bf16(false, a, false, b2,
                                                 (short)0, acc[g * 4 + 2], false, false);
        acc[g * 4 + 3] = __builtin_amdgcn_wmma_f32_16x16x32_bf16(false, a, false, b3,
                                                 (short)0, acc[g * 4 + 3], false, false);
      }
    }
    __syncthreads();
  }

  // ---- Epilogue: y += combine * (acc + b_out); skip tokens with combine==0 ----
  float cw[8];
#pragma unroll
  for (int v = 0; v < 8; ++v)
    cw[v] = comb[(size_t)(tbase + mg2 * 16 + v + 8 * hi) * E_NUM + e];

#pragma unroll
  for (int v = 0; v < 8; ++v) {
    if (cw[v] == 0.f) continue;
    const size_t rowoff = (size_t)(tbase + mg2 * 16 + v + 8 * hi) * D_DIM;
    const float c = cw[v];
#pragma unroll 4
    for (int nt = 0; nt < 16; ++nt) {
      int d = nq * 256 + nt * 16 + l15;
      float val = c * (acc[nt][v] + bout[(size_t)e * D_DIM + d]);
      atomicAdd(out + rowoff + d, val);
    }
  }
}

// ---------------------------------------------------------------------------
extern "C" void kernel_launch(void* const* d_in, const int* in_sizes, int n_in,
                              void* d_out, int out_size, void* d_ws, size_t ws_size,
                              hipStream_t stream) {
  const float* x    = (const float*)d_in[0];  // residual [B,S,D]
  const float* wr   = (const float*)d_in[1];  // W_router [E,D]
  const float* win  = (const float*)d_in[2];  // W_in     [E,D,F]
  const float* bin  = (const float*)d_in[3];  // b_in     [E,F]
  const float* wout = (const float*)d_in[4];  // W_out    [E,F,D]
  const float* bout = (const float*)d_in[5];  // b_out    [E,D]
  float* out = (float*)d_out;

  const int T = in_sizes[0] / D_DIM;          // B*S = 8192
  float* comb = (float*)d_ws;                 // [T, E] combine weights

  hipMemsetAsync(d_out, 0, (size_t)T * D_DIM * sizeof(float), stream);

  moe_router<<<dim3((T + 7) / 8), dim3(256), 0, stream>>>(x, wr, comb, T);

  moe_expert<<<dim3(T / TILE_T, E_NUM), dim3(256),
               LDS_HALVES * sizeof(__bf16), stream>>>(x, win, bin, wout, bout,
                                                      comb, out);
}
